// ContrastiveLoss_137438953481
// MI455X (gfx1250) — matrix operand order repaired and checked by reading, not compiled, not run on hardware
//
#include <hip/hip_runtime.h>
#include <hip/hip_bf16.h>

// CDNA5 / gfx1250 wave32 WMMA types (probe-confirmed signatures)
typedef __attribute__((ext_vector_type(16))) __bf16 v16bf;
typedef __attribute__((ext_vector_type(8)))  float  v8f;
typedef __attribute__((ext_vector_type(4)))  __bf16 v4bf;

#define B_ROWS 8192
#define DIM    2048
#define TILE   128
#define KC     32
#define LDSPITCH (KC + 8)               // padded row stride (80 B) for bank spread
#define NT     (B_ROWS / TILE)          // 64 tile rows/cols
#define NBLK   (NT * (NT + 1) / 2)      // 2080 upper-triangular tile pairs
#define MARGIN 2.0f

// ---- gfx1250 async global->LDS copy (ASYNCcnt-tracked DMA, ISA ch.10) ----
// LDS operand is a byte offset in a VGPR; generic LDS pointer low 32 bits are
// exactly that offset.
__device__ __forceinline__ void async_copy_b128(void* lds, const void* gptr) {
  uint32_t loff = (uint32_t)(uintptr_t)lds;
  asm volatile("global_load_async_to_lds_b128 %0, %1, off"
               :: "v"(loff), "v"(gptr) : "memory");
}
__device__ __forceinline__ void wait_async_le8() {
  asm volatile("s_wait_asynccnt 8" ::: "memory");
}
__device__ __forceinline__ void wait_async_0() {
  asm volatile("s_wait_asynccnt 0" ::: "memory");
}

// -------- Kernel 1: bf16 hi/lo split of P (done ONCE) + row squared norms ----
// One wave per row. p = hi + lo with hi = bf16(p), lo = bf16(p - f32(hi)):
// dot(p,q) ~= hi*hi + hi*lo + lo*hi at bf16 matrix-core rate, ~f32 accuracy.
__global__ __launch_bounds__(256)
void split_rowsq_kernel(const float* __restrict__ p,
                        __bf16* __restrict__ phi,
                        __bf16* __restrict__ plo,
                        float* __restrict__ sq) {
  int wave = threadIdx.x >> 5;
  int lane = threadIdx.x & 31;
  int row  = blockIdx.x * 8 + wave;
  const float4* pr = (const float4*)(p + (size_t)row * DIM);
  v4bf* hr = (v4bf*)(phi + (size_t)row * DIM);
  v4bf* lr = (v4bf*)(plo + (size_t)row * DIM);
  float s = 0.f;
#pragma unroll
  for (int it = 0; it < DIM / 4 / 32; ++it) {        // 16 float4 per lane
    int idx = lane + it * 32;
    float4 v = pr[idx];
    float a[4] = {v.x, v.y, v.z, v.w};
    v4bf h, l;
#pragma unroll
    for (int e = 0; e < 4; ++e) {
      s += a[e] * a[e];
      __bf16 hh = (__bf16)a[e];
      h[e] = hh;
      l[e] = (__bf16)(a[e] - (float)hh);
    }
    hr[idx] = h;
    lr[idx] = l;
  }
  for (int off = 16; off > 0; off >>= 1) s += __shfl_down(s, off, 32);
  if (lane == 0) sq[row] = s;
}

// -------- Kernel 2: fused symmetric-GEMM + contrastive epilogue --------
// One block per (ti, tj) tile pair, tj >= ti (symmetry: 2x less work).
// 256 threads = 8 waves; each wave owns a 64x32 sub-tile = 4x2 WMMA tiles.
// Panels are DMA'd global->LDS with async b128 copies, double-buffered so the
// next K-chunk's DMA overlaps this chunk's v_wmma issue.
__global__ __launch_bounds__(256)
void tile_loss_kernel(const __bf16* __restrict__ phi,
                      const __bf16* __restrict__ plo,
                      const long long* __restrict__ gt,
                      const float* __restrict__ sq,
                      float* __restrict__ partial) {
  __shared__ __align__(16) __bf16 Ahi[2][TILE][LDSPITCH], Alo[2][TILE][LDSPITCH];
  __shared__ __align__(16) __bf16 Bhi[2][TILE][LDSPITCH], Blo[2][TILE][LDSPITCH];
  __shared__ float sqI[TILE], sqJ[TILE];
  __shared__ int   gI[TILE], gJ[TILE];
  __shared__ float red[8];

  // Decode linear block id -> upper-triangular (ti, tj)
  int L = blockIdx.x;
  int ti = 0, rem = L;
  while (rem >= NT - ti) { rem -= NT - ti; ++ti; }
  int tj = ti + rem;
  int i0 = ti * TILE, j0 = tj * TILE;

  int tid  = threadIdx.x;
  int wave = tid >> 5, lane = tid & 31;
  int wr = wave >> 2, wc = wave & 3;   // 2x4 wave grid over the 128x128 tile
  int lrow   = lane & 15;
  int hihalf = lane >> 4;

  if (tid < TILE) { sqI[tid] = sq[i0 + tid]; gI[tid] = (int)gt[i0 + tid]; }
  else { int t = tid - TILE; sqJ[t] = sq[j0 + t]; gJ[t] = (int)gt[j0 + t]; }

  // Per-thread staging slots: 512 transfers of 16B per panel-set, 2 per thread.
  // row = u>>2, element offset c8 = (u&3)*8 (8 bf16 = 16 bytes).
  int u0 = tid, u1 = tid + 256;
  int sr0 = u0 >> 2, sc0 = (u0 & 3) * 8;
  int sr1 = u1 >> 2, sc1 = (u1 & 3) * 8;

  // Issue one K-chunk's 4 panels (Ahi/Alo/Bhi/Blo): 8 async instrs per thread.
#define STAGE(buf, kc)                                                          \
  do {                                                                          \
    const __bf16* aih = phi + (size_t)(i0 + sr0) * DIM + (kc) + sc0;            \
    async_copy_b128(&Ahi[buf][sr0][sc0], aih);                                  \
    async_copy_b128(&Alo[buf][sr0][sc0], plo + (size_t)(i0 + sr0) * DIM + (kc) + sc0); \
    async_copy_b128(&Bhi[buf][sr0][sc0], phi + (size_t)(j0 + sr0) * DIM + (kc) + sc0); \
    async_copy_b128(&Blo[buf][sr0][sc0], plo + (size_t)(j0 + sr0) * DIM + (kc) + sc0); \
    async_copy_b128(&Ahi[buf][sr1][sc1], phi + (size_t)(i0 + sr1) * DIM + (kc) + sc1); \
    async_copy_b128(&Alo[buf][sr1][sc1], plo + (size_t)(i0 + sr1) * DIM + (kc) + sc1); \
    async_copy_b128(&Bhi[buf][sr1][sc1], phi + (size_t)(j0 + sr1) * DIM + (kc) + sc1); \
    async_copy_b128(&Blo[buf][sr1][sc1], plo + (size_t)(j0 + sr1) * DIM + (kc) + sc1); \
  } while (0)

  v8f acc[4][2] = {};

  STAGE(0, 0);                                   // prologue: chunk 0 -> buf 0

  for (int kc = 0; kc < DIM; kc += KC) {
    int buf = (kc / KC) & 1;
    bool pre = (kc + KC) < DIM;
    if (pre) STAGE(buf ^ 1, kc + KC);            // prefetch next chunk
    if (pre) wait_async_le8();                   // oldest 8 (this buf) done
    else     wait_async_0();
    __syncthreads();                             // panels visible block-wide

    // Fragment loads in contiguous k order (k = 16*hihalf + s), same for A and B.
    // B (32x16): lane = col n (lane%16), K = hihalf?16..31:0..15, contiguous 32B.
    v16bf bfh[2], bfl[2];
#pragma unroll
    for (int n = 0; n < 2; ++n) {
      int r  = wc * 32 + n * 16 + lrow;
      int kk = hihalf * 16;
      __builtin_memcpy(&bfh[n], &Bhi[buf][r][kk], 32);
      __builtin_memcpy(&bfl[n], &Blo[buf][r][kk], 32);
    }
    // A (16x32): lane = row m (lane%16), K = hihalf?16..31:0..15, contiguous 32B.
    v16bf afh[4], afl[4];
#pragma unroll
    for (int m = 0; m < 4; ++m) {
      int r  = wr * 64 + m * 16 + lrow;
      int kk = hihalf * 16;
      __builtin_memcpy(&afh[m], &Ahi[buf][r][kk], 32);
      __builtin_memcpy(&afl[m], &Alo[buf][r][kk], 32);
    }

#pragma unroll
    for (int m = 0; m < 4; ++m)
#pragma unroll
      for (int n = 0; n < 2; ++n) {
        acc[m][n] = __builtin_amdgcn_wmma_f32_16x16x32_bf16(
            false, afh[m], false, bfh[n], (short)0, acc[m][n], false, false);
        acc[m][n] = __builtin_amdgcn_wmma_f32_16x16x32_bf16(
            false, afh[m], false, bfl[n], (short)0, acc[m][n], false, false);
        acc[m][n] = __builtin_amdgcn_wmma_f32_16x16x32_bf16(
            false, afl[m], false, bfh[n], (short)0, acc[m][n], false, false);
      }

    __syncthreads();   // all waves done reading buf before next overwrite
  }
#undef STAGE

  // Epilogue: C layout (16x16 f32): VGPR v -> M = v + 8*hihalf, N = lane%16.
  float local = 0.f;
#pragma unroll
  for (int m = 0; m < 4; ++m)
#pragma unroll
    for (int n = 0; n < 2; ++n)
#pragma unroll
      for (int v = 0; v < 8; ++v) {
        int il = wr * 64 + m * 16 + v + hihalf * 8;
        int jl = wc * 32 + n * 16 + lrow;
        float dot = acc[m][n][v];
        float d2  = fmaxf(sqI[il] + sqJ[jl] - 2.0f * dot, 0.0f) * (1.0f / DIM);
        float term = (gI[il] == gJ[jl]) ? d2 : fmaxf(MARGIN - d2, 0.0f);
        bool inc = (ti != tj) || (jl > il);   // strict upper triangle
        local += inc ? term : 0.0f;
      }

  for (int off = 16; off > 0; off >>= 1) local += __shfl_down(local, off, 32);
  if (lane == 0) red[wave] = local;
  __syncthreads();
  if (tid == 0) {
    float s = 0.f;
#pragma unroll
    for (int w = 0; w < 8; ++w) s += red[w];
    partial[blockIdx.x] = s;
  }
}

// -------- Kernel 3: deterministic final reduction + scaling --------
__global__ __launch_bounds__(256)
void final_reduce_kernel(const float* __restrict__ partial, float* __restrict__ out) {
  __shared__ float red[256];
  float s = 0.f;
  for (int i = threadIdx.x; i < NBLK; i += 256) s += partial[i];
  red[threadIdx.x] = s;
  __syncthreads();
  for (int off = 128; off > 0; off >>= 1) {
    if (threadIdx.x < off) red[threadIdx.x] += red[threadIdx.x + off];
    __syncthreads();
  }
  if (threadIdx.x == 0)
    out[0] = 2.0f * red[0] / ((float)B_ROWS * (float)(B_ROWS - 1));
}

extern "C" void kernel_launch(void* const* d_in, const int* in_sizes, int n_in,
                              void* d_out, int out_size, void* d_ws, size_t ws_size,
                              hipStream_t stream) {
  (void)in_sizes; (void)n_in; (void)out_size; (void)ws_size;
  const float*     p  = (const float*)d_in[0];
  const long long* gt = (const long long*)d_in[1];

  // Workspace layout: Phi (32MB) | Plo (32MB) | sq (32KB) | partials (8.3KB)
  __bf16* phi = (__bf16*)d_ws;
  __bf16* plo = phi + (size_t)B_ROWS * DIM;
  float*  sq  = (float*)(plo + (size_t)B_ROWS * DIM);
  float*  partial = sq + B_ROWS;

  split_rowsq_kernel<<<B_ROWS / 8, 256, 0, stream>>>(p, phi, plo, sq);
  tile_loss_kernel<<<NBLK, 256, 0, stream>>>(phi, plo, gt, sq, partial);
  final_reduce_kernel<<<1, 256, 0, stream>>>(partial, (float*)d_out);
}
